// GCNconv_53858889892141
// MI455X (gfx1250) — compile-verified, matrix operand order checked by reading, not run on hardware
//
#include <hip/hip_runtime.h>
#include <hip/hip_bf16.h>
#include <math.h>

typedef __attribute__((ext_vector_type(16))) _Float16 v16h;
typedef __attribute__((ext_vector_type(8)))  float    v8f;

#define NPG 784

// ---------- degree / conv1 scalar path ----------
__global__ void init_deg_s(float* __restrict__ deg, float* __restrict__ s, int n) {
  int i = blockIdx.x * blockDim.x + threadIdx.x;
  if (i < n) { deg[i] = 1.0f; s[i] = 0.0f; }   // +1 = self loop
}

__global__ void deg_scatter(const int* __restrict__ ei, float* __restrict__ deg, long long E) {
  long long e = (long long)blockIdx.x * blockDim.x + threadIdx.x;
  if (e < E) { int dst = ei[E + e]; atomicAdd(&deg[dst], 1.0f); }
}

__global__ void dinv_k(const float* __restrict__ deg, float* __restrict__ dinv, int n) {
  int i = blockIdx.x * blockDim.x + threadIdx.x;
  if (i < n) dinv[i] = rsqrtf(deg[i]);
}

__global__ void s_scatter(const int* __restrict__ ei, const float* __restrict__ x,
                          const float* __restrict__ dinv, float* __restrict__ s, long long E) {
  long long e = (long long)blockIdx.x * blockDim.x + threadIdx.x;
  if (e < E) {
    int src = ei[e], dst = ei[E + e];
    atomicAdd(&s[dst], x[src] * dinv[src] * dinv[dst]);
  }
}

// h1t[n][c] = tanh( W1[c]*(s[n] + x[n]*dinv^2) + b1[c] ), stored f16 for WMMA
__global__ void h1_k(const float* __restrict__ x, const float* __restrict__ s,
                     const float* __restrict__ dinv, const float* __restrict__ W1,
                     const float* __restrict__ b1, _Float16* __restrict__ h1t, long long total) {
  long long t = (long long)blockIdx.x * blockDim.x + threadIdx.x;
  if (t < total) {
    int node = (int)(t >> 6); int c = (int)(t & 63);
    float di = dinv[node];
    float v = fmaf(W1[c], s[node] + x[node] * di * di, b1[c]);
    h1t[t] = (_Float16)tanhf(v);
  }
}

// W2t[n][k] = W2[k][n]  (f16, K-contiguous so B fragments vectorize like A)
__global__ void cvt_w2t(const float* __restrict__ W2, _Float16* __restrict__ W2t) {
  int i = blockIdx.x * blockDim.x + threadIdx.x;
  if (i < 64 * 128) {
    int k = i >> 7, n = i & 127;
    W2t[n * 64 + k] = (_Float16)W2[i];
  }
}

// ---------- conv2 dense part: [N x 64] @ [64 x 128] via v_wmma_f32_16x16x32_f16 ----------
// Block = 256 threads = 8 waves, covers 128 rows. Wave w owns M-tile w (16 rows),
// keeps its A fragments in registers, and loops over all 8 N-tiles -> 16 WMMA/wave.
__global__ void gemm_wmma(const _Float16* __restrict__ A, const _Float16* __restrict__ Bt,
                          _Float16* __restrict__ C) {
  __shared__ _Float16 As[128 * 64];  // 128 rows x K=64, 16KB
  __shared__ _Float16 Bs[128 * 64];  // W2^T: 128 cols x K=64, 16KB
  long long m0 = (long long)blockIdx.x * 128;

  for (int i = threadIdx.x * 8; i < 128 * 64; i += 256 * 8) {   // 16B vector copies
    *(float4*)(&As[i]) = *(const float4*)(&A[m0 * 64 + i]);
    *(float4*)(&Bs[i]) = *(const float4*)(&Bt[i]);
  }
  __syncthreads();

  const int lane = threadIdx.x & 31;
  const int wv   = threadIdx.x >> 5;
  const int mr   = wv * 16;        // this wave's M-tile base (local)
  const int ml   = lane & 15;      // row (A frag) / col (B frag)
  const int half = lane >> 4;      // K-octet selector per ISA layout

  // A fragments once, reused across all 8 N-tiles
  v16h a0, a1;
#pragma unroll
  for (int i = 0; i < 16; ++i) {
    int kbase = ((i < 8) ? 0 : 16) + half * 8 + (i & 7);   // contiguous octets -> ds_load_b128
    a0[i] = As[(mr + ml) * 64 + kbase];
    a1[i] = As[(mr + ml) * 64 + 32 + kbase];
  }

#pragma unroll
  for (int nt = 0; nt < 8; ++nt) {
    const int n0 = nt * 16;
    v16h b0, b1v;
#pragma unroll
    for (int i = 0; i < 16; ++i) {
      int kbase = ((i < 8) ? 0 : 16) + half * 8 + (i & 7);
      b0[i]  = Bs[(n0 + ml) * 64 + kbase];
      b1v[i] = Bs[(n0 + ml) * 64 + 32 + kbase];
    }
    v8f c = {};
    c = __builtin_amdgcn_wmma_f32_16x16x32_f16(false, a0, false, b0,  (short)0, c, false, false);
    c = __builtin_amdgcn_wmma_f32_16x16x32_f16(false, a1, false, b1v, (short)0, c, false, false);
    // D layout: n = n0 + (lane&15), m = r + 8*half
#pragma unroll
    for (int r = 0; r < 8; ++r) {
      long long m = m0 + mr + half * 8 + r;
      C[m * 128 + n0 + ml] = (_Float16)c[r];
    }
  }
}

// agg2 seeded with self-loop message + bias: agg2 = h2lin*dinv^2 + b2
__global__ void agg2_init(const _Float16* __restrict__ h2lin, const float* __restrict__ dinv,
                          const float* __restrict__ b2, float* __restrict__ agg2, long long total) {
  long long t = (long long)blockIdx.x * blockDim.x + threadIdx.x;
  if (t < total) {
    int node = (int)(t >> 7); int c = (int)(t & 127);
    float di = dinv[node];
    agg2[t] = fmaf((float)h2lin[t], di * di, b2[c]);
  }
}

// layer-2 edge scatter: 32 threads/edge, 4 channels each (the bandwidth-dominant kernel)
__global__ void e2_scatter(const int* __restrict__ ei, const _Float16* __restrict__ h2lin,
                           const float* __restrict__ dinv, float* __restrict__ agg2, long long E) {
  long long t = (long long)blockIdx.x * blockDim.x + threadIdx.x;
  if (t < E * 32) {
    long long e = t >> 5; int c0 = (int)(t & 31) * 4;
    int src = ei[e], dst = ei[E + e];
    float norm = dinv[src] * dinv[dst];
    const _Float16* hp = h2lin + (long long)src * 128 + c0;   // 8B contiguous -> b64 load
    float* ap = agg2 + (long long)dst * 128 + c0;
#pragma unroll
    for (int j = 0; j < 4; ++j) atomicAdd(&ap[j], (float)hp[j] * norm);
  }
}

__global__ void zero_pooled(float* __restrict__ pooled, int n) {
  int i = blockIdx.x * blockDim.x + threadIdx.x;
  if (i < n) pooled[i] = 0.0f;
}

// tanh finalize fused with mean-pool (h2 never materialized)
__global__ void pool_k(const float* __restrict__ agg2, float* __restrict__ pooled, long long total) {
  long long t = (long long)blockIdx.x * blockDim.x + threadIdx.x;
  if (t < total) {
    float v = tanhf(agg2[t]);
    int node = (int)(t >> 7); int c = (int)(t & 127); int g = node / NPG;
    atomicAdd(&pooled[(long long)g * 128 + c], v * (1.0f / (float)NPG));
  }
}

// per-graph heads + fusion + log_softmax; one block of 128 threads per graph
__global__ void head_k(const float* __restrict__ x, const float* __restrict__ pooled,
                       const float* __restrict__ Wfc, const float* __restrict__ bfc,
                       const float* __restrict__ Wr1, const float* __restrict__ br1,
                       const float* __restrict__ Wr2, const float* __restrict__ br2,
                       const float* __restrict__ fw, float* __restrict__ out) {
  int g = blockIdx.x; int c = threadIdx.x;          // 128 threads
  __shared__ float r1[128];
  __shared__ float pl[128];
  __shared__ float fused[10];
  const float* xg = x + (long long)g * NPG;
  float acc = br1[c];
  for (int i = 0; i < NPG; ++i) acc = fmaf(xg[i], Wr1[i * 128 + c], acc);
  r1[c] = fmaxf(acc, 0.0f);
  pl[c] = pooled[(long long)g * 128 + c];
  __syncthreads();
  if (c < 10) {
    float lg = bfc[c], rr = br2[c];
    for (int k = 0; k < 128; ++k) {
      lg = fmaf(pl[k], Wfc[k * 10 + c], lg);
      rr = fmaf(r1[k], Wr2[k * 10 + c], rr);
    }
    float sw = 1.0f / (1.0f + expf(-fw[0]));
    fused[c] = sw * lg + (1.0f - sw) * rr;
  }
  __syncthreads();
  if (c < 10) {
    float mx = fused[0];
    for (int j = 1; j < 10; ++j) mx = fmaxf(mx, fused[j]);
    float se = 0.0f;
    for (int j = 0; j < 10; ++j) se += expf(fused[j] - mx);
    out[(long long)g * 10 + c] = fused[c] - mx - logf(se);
  }
}

extern "C" void kernel_launch(void* const* d_in, const int* in_sizes, int n_in,
                              void* d_out, int out_size, void* d_ws, size_t ws_size,
                              hipStream_t stream) {
  const float* x    = (const float*)d_in[0];
  const int*   ei   = (const int*)d_in[1];      // [2, E]
  const float* W1   = (const float*)d_in[3];
  const float* b1   = (const float*)d_in[4];
  const float* W2   = (const float*)d_in[5];
  const float* b2   = (const float*)d_in[6];
  const float* Wfc  = (const float*)d_in[7];
  const float* bfc  = (const float*)d_in[8];
  const float* Wr1  = (const float*)d_in[9];
  const float* br1  = (const float*)d_in[10];
  const float* Wr2  = (const float*)d_in[11];
  const float* br2  = (const float*)d_in[12];
  const float* fw   = (const float*)d_in[13];
  float* out = (float*)d_out;

  const int       N = in_sizes[0];
  const long long E = (long long)in_sizes[1] / 2;
  const int       G = N / NPG;

  // workspace carve-out (256B aligned)
  size_t off = 0;
  auto carve = [&](size_t bytes) -> char* {
    char* p = (char*)d_ws + off;
    off += (bytes + 255) & ~(size_t)255;
    return p;
  };
  float*    deg    = (float*)   carve((size_t)N * 4);
  float*    dinv   = (float*)   carve((size_t)N * 4);
  float*    s      = (float*)   carve((size_t)N * 4);
  _Float16* h1t    = (_Float16*)carve((size_t)N * 64 * 2);
  _Float16* W2t    = (_Float16*)carve((size_t)64 * 128 * 2);
  _Float16* h2lin  = (_Float16*)carve((size_t)N * 128 * 2);
  float*    agg2   = (float*)   carve((size_t)N * 128 * 4);
  float*    pooled = (float*)   carve((size_t)G * 128 * 4);
  (void)ws_size; (void)n_in; (void)out_size;

  const int B = 256;
  auto blocks = [&](long long work) -> unsigned { return (unsigned)((work + B - 1) / B); };

  init_deg_s <<<blocks(N), B, 0, stream>>>(deg, s, N);
  deg_scatter<<<blocks(E), B, 0, stream>>>(ei, deg, E);
  dinv_k     <<<blocks(N), B, 0, stream>>>(deg, dinv, N);
  s_scatter  <<<blocks(E), B, 0, stream>>>(ei, x, dinv, s, E);
  h1_k       <<<blocks((long long)N * 64), B, 0, stream>>>(x, s, dinv, W1, b1, h1t, (long long)N * 64);
  cvt_w2t    <<<blocks(64 * 128), B, 0, stream>>>(W2, W2t);
  gemm_wmma  <<<(unsigned)(N / 128), B, 0, stream>>>(h1t, W2t, h2lin);
  agg2_init  <<<blocks((long long)N * 128), B, 0, stream>>>(h2lin, dinv, b2, agg2, (long long)N * 128);
  e2_scatter <<<blocks(E * 32), B, 0, stream>>>(ei, h2lin, dinv, agg2, E);
  zero_pooled<<<blocks(G * 128), B, 0, stream>>>(pooled, G * 128);
  pool_k     <<<blocks((long long)N * 128), B, 0, stream>>>(agg2, pooled, (long long)N * 128);
  head_k     <<<(unsigned)G, 128, 0, stream>>>(x, pooled, Wfc, bfc, Wr1, br1, Wr2, br2, fw, out);
}
